// GPT2AttentionXWWX_20572893347949
// MI455X (gfx1250) — matrix operand order repaired and checked by reading, not compiled, not run on hardware
//
#include <hip/hip_runtime.h>
#include <cstdint>
#include <cstddef>

// ---------------------------------------------------------------------------
// Types for CDNA5 WMMA (wave32): 16x16x32 bf16 -> f32
// ---------------------------------------------------------------------------
typedef __attribute__((ext_vector_type(16))) __bf16 v16bf;
typedef __attribute__((ext_vector_type(8)))  float  v8f;

union FragAB { uint4 q[2]; v16bf v; };      // 32 bytes: one A or B fragment
union U4U16  { uint4 q; unsigned short h[8]; };

__device__ __forceinline__ unsigned short f2bf(float f) {
  union { float f; uint32_t u; } x; x.f = f;
  uint32_t u = x.u;
  return (unsigned short)((u + 0x7FFFu + ((u >> 16) & 1u)) >> 16);  // RNE
}

// ---------------------------------------------------------------------------
// fp32 -> bf16 conversion / re-layout kernels (one-shot, bandwidth-trivial)
// ---------------------------------------------------------------------------
__global__ void cvt_f32_bf16_kernel(const float* __restrict__ in,
                                    unsigned short* __restrict__ out, int n) {
  int i = blockIdx.x * blockDim.x + threadIdx.x;
  if (i < n) out[i] = f2bf(in[i]);
}

// out[n*Kd + k] = in[k*ld + off + n]   (store B^T layout [N][K] from [K][N] slice)
__global__ void cvt_transpose_kernel(const float* __restrict__ in,
                                     unsigned short* __restrict__ out,
                                     int N, int Kd, int ld, int off) {
  int i = blockIdx.x * blockDim.x + threadIdx.x;
  if (i >= N * Kd) return;
  int n = i / Kd, k = i % Kd;
  out[i] = f2bf(in[(size_t)k * ld + off + n]);
}

// out[n*Kd + k] = in[n*ld + off + k]   (strided row copy; used for Wk since
// q = t @ Wk^T means B^T layout == Wk itself)
__global__ void cvt_rows_kernel(const float* __restrict__ in,
                                unsigned short* __restrict__ out,
                                int N, int Kd, int ld, int off) {
  int i = blockIdx.x * blockDim.x + threadIdx.x;
  if (i >= N * Kd) return;
  int n = i / Kd, k = i % Kd;
  out[i] = f2bf(in[(size_t)n * ld + off + k]);
}

// ---------------------------------------------------------------------------
// Tiled bf16 GEMM:  C[M,N] = A[M,K] * B[K,N] + bias[N]
//   A: bf16 row-major.  BT: bf16 [N][K] (i.e. B transposed, rows contiguous).
//   Block 256 thr (8 waves), tile 128x128, K-step 32, wave tile 64x32 (4x2 WMMA).
// ---------------------------------------------------------------------------
template<bool F32OUT>
__global__ __launch_bounds__(256)
void gemm_bf16_kernel(const unsigned short* __restrict__ A,
                      const unsigned short* __restrict__ BT,
                      const float* __restrict__ bias,
                      unsigned short* __restrict__ Cbf,
                      float* __restrict__ Cf,
                      int M, int N, int K) {
  constexpr int BM = 128, BN = 128, BK = 32;
  __shared__ unsigned short Asm[BM][BK];
  __shared__ unsigned short Bsm[BN][BK];

  const int t    = threadIdx.x;
  const int lane = t & 31;
  const int wave = t >> 5;
  const int wm   = (wave & 1) * 64;   // wave M offset inside block tile
  const int wn   = (wave >> 1) * 32;  // wave N offset inside block tile
  const int m0   = blockIdx.y * BM;
  const int n0   = blockIdx.x * BN;
  const int half = lane >> 4;         // 0: lanes 0-15, 1: lanes 16-31
  const int l15  = lane & 15;

  v8f acc[4][2];
  #pragma unroll
  for (int i = 0; i < 4; ++i)
    #pragma unroll
    for (int j = 0; j < 2; ++j)
      #pragma unroll
      for (int r = 0; r < 8; ++r) acc[i][j][r] = 0.0f;

  for (int kb = 0; kb < K; kb += BK) {
    __syncthreads();
    // Stage A and B tiles: 512 chunks of 16B each, 2 per thread.
    #pragma unroll
    for (int c = 0; c < 2; ++c) {
      int id  = t + c * 256;
      int row = id >> 2;            // 4 x 16B chunks per 32-elem row
      int col = (id & 3) * 8;
      *(uint4*)&Asm[row][col] = *(const uint4*)&A [(size_t)(m0 + row) * K + kb + col];
      *(uint4*)&Bsm[row][col] = *(const uint4*)&BT[(size_t)(n0 + row) * K + kb + col];
    }
    if (kb + BK < K) {              // prefetch next K tile -> global_prefetch_b8
      int row = t >> 2, col = (t & 3) * 8;
      __builtin_prefetch(&A [(size_t)(m0 + row) * K + kb + BK + col], 0, 0);
      __builtin_prefetch(&BT[(size_t)(n0 + row) * K + kb + BK + col], 0, 0);
    }
    __syncthreads();

    // A fragments: lane half 0 holds K 0-7 & 16-23, half 1 holds K 8-15 & 24-31.
    FragAB a[4], b[2];
    #pragma unroll
    for (int i = 0; i < 4; ++i) {
      int row = wm + i * 16 + l15;
      int klo = half ? 8 : 0;
      a[i].q[0] = *(const uint4*)&Asm[row][klo];
      a[i].q[1] = *(const uint4*)&Asm[row][klo + 16];
    }
    // B fragments: lane = column; half 0 holds K 0-7 & 16-23, half 1 holds K 8-15 & 24-31.
    #pragma unroll
    for (int j = 0; j < 2; ++j) {
      int col = wn + j * 16 + l15;
      int klo = half ? 8 : 0;
      b[j].q[0] = *(const uint4*)&Bsm[col][klo];
      b[j].q[1] = *(const uint4*)&Bsm[col][klo + 16];
    }

    #pragma unroll
    for (int i = 0; i < 4; ++i)
      #pragma unroll
      for (int j = 0; j < 2; ++j)
        acc[i][j] = __builtin_amdgcn_wmma_f32_16x16x32_bf16(
            false, a[i].v, false, b[j].v, (short)0, acc[i][j], false, false);
  }

  // Epilogue: C/D layout -> VGPR r holds row r (half 0) or r+8 (half 1).
  #pragma unroll
  for (int i = 0; i < 4; ++i) {
    #pragma unroll
    for (int j = 0; j < 2; ++j) {
      int n  = n0 + wn + j * 16 + l15;
      float bv = bias[n];
      #pragma unroll
      for (int r = 0; r < 8; ++r) {
        int mm = m0 + wm + i * 16 + (half ? r + 8 : r);
        float v = acc[i][j][r] + bv;
        if (F32OUT) Cf [(size_t)mm * N + n] = v;
        else        Cbf[(size_t)mm * N + n] = f2bf(v);
      }
    }
  }
}

// ---------------------------------------------------------------------------
// Flash attention (causal), one block per (b, h, 128 query rows).
//   Q: [B,S,E] bf16 (fused query), K source: raw x bf16, V: bf16.
//   8 waves * 16 q-rows; KV tiles of 64; D = 64 (2 WMMA K-steps).
// ---------------------------------------------------------------------------
__global__ __launch_bounds__(256)
void flash_attn_kernel(const unsigned short* __restrict__ Q,
                       const unsigned short* __restrict__ Kx,
                       const unsigned short* __restrict__ V,
                       unsigned short* __restrict__ O,
                       int S, int E, float scale) {
  constexpr int D = 64, BQ = 128, BKV = 64;
  __shared__ unsigned short Ksm[BKV][D];        // keys, row-major
  __shared__ unsigned short Vsm[D][BKV];        // values, transposed
  __shared__ unsigned short Psm[8][16][BKV];    // per-wave P staging

  const int t    = threadIdx.x;
  const int lane = t & 31;
  const int wave = t >> 5;
  const int half = lane >> 4;
  const int l15  = lane & 15;
  const int q0   = blockIdx.x * BQ;
  const int h    = blockIdx.y;
  const int b    = blockIdx.z;
  const size_t base = (size_t)b * S * E + (size_t)h * D;

  // Load the wave's Q fragments once (2 K-steps over D=64).
  FragAB qa[2];
  {
    int qi = q0 + wave * 16 + l15;
    const unsigned short* qrow = Q + base + (size_t)qi * E;
    #pragma unroll
    for (int kk = 0; kk < 2; ++kk) {
      int klo = kk * 32 + (half ? 8 : 0);
      qa[kk].q[0] = *(const uint4*)&qrow[klo];
      qa[kk].q[1] = *(const uint4*)&qrow[klo + 16];
    }
  }

  v8f oacc[4];
  #pragma unroll
  for (int j = 0; j < 4; ++j)
    #pragma unroll
    for (int r = 0; r < 8; ++r) oacc[j][r] = 0.0f;
  float mrow[8], lrow[8];
  #pragma unroll
  for (int r = 0; r < 8; ++r) { mrow[r] = -1e30f; lrow[r] = 0.0f; }

  const int nkv = (q0 + BQ) / BKV;          // causal: only tiles at/below diag
  for (int cb = 0; cb < nkv; ++cb) {
    const int c0 = cb * BKV;
    __syncthreads();
    // Stage K (row-major) and V (transposed): 512 x 16B chunks, 2 per thread.
    #pragma unroll
    for (int c = 0; c < 2; ++c) {
      int id = t + c * 256;
      int kv = id >> 3;
      int d0 = (id & 7) * 8;
      *(uint4*)&Ksm[kv][d0] = *(const uint4*)&Kx[base + (size_t)(c0 + kv) * E + d0];
      U4U16 u; u.q = *(const uint4*)&V[base + (size_t)(c0 + kv) * E + d0];
      #pragma unroll
      for (int j2 = 0; j2 < 8; ++j2) Vsm[d0 + j2][kv] = u.h[j2];
    }
    __syncthreads();

    // ---- scores: S = Q @ K^T  (B^T of K^T is K itself -> contiguous reads)
    v8f sacc[4];
    #pragma unroll
    for (int nf = 0; nf < 4; ++nf) {
      #pragma unroll
      for (int r = 0; r < 8; ++r) sacc[nf][r] = 0.0f;
      #pragma unroll
      for (int kk = 0; kk < 2; ++kk) {
        FragAB bfr;
        int key = nf * 16 + l15;
        int klo = kk * 32 + (half ? 8 : 0);
        bfr.q[0] = *(const uint4*)&Ksm[key][klo];
        bfr.q[1] = *(const uint4*)&Ksm[key][klo + 16];
        sacc[nf] = __builtin_amdgcn_wmma_f32_16x16x32_bf16(
            false, qa[kk].v, false, bfr.v, (short)0, sacc[nf], false, false);
      }
    }

    // ---- causal mask + online softmax (fp32, per-row stats in 16-lane halves)
    float p[4][8];
    #pragma unroll
    for (int r = 0; r < 8; ++r) {
      int qi = q0 + wave * 16 + (half ? r + 8 : r);
      float rmax = -1e30f;
      #pragma unroll
      for (int nf = 0; nf < 4; ++nf) {
        int ki = c0 + nf * 16 + l15;
        float s = sacc[nf][r] * scale;
        if (ki > qi) s = -1e30f;
        p[nf][r] = s;
        rmax = fmaxf(rmax, s);
      }
      #pragma unroll
      for (int off = 1; off < 16; off <<= 1)
        rmax = fmaxf(rmax, __shfl_xor(rmax, off, 32));
      float mnew  = fmaxf(mrow[r], rmax);
      float alpha = __expf(mrow[r] - mnew);
      float rsum  = 0.0f;
      #pragma unroll
      for (int nf = 0; nf < 4; ++nf) {
        float e = __expf(p[nf][r] - mnew);
        p[nf][r] = e;
        rsum += e;
      }
      #pragma unroll
      for (int off = 1; off < 16; off <<= 1)
        rsum += __shfl_xor(rsum, off, 32);
      lrow[r] = lrow[r] * alpha + rsum;
      mrow[r] = mnew;
      #pragma unroll
      for (int j = 0; j < 4; ++j) oacc[j][r] *= alpha;
    }

    // ---- P (C-layout) -> LDS -> A-layout fragments; O += P @ V
    #pragma unroll
    for (int r = 0; r < 8; ++r) {
      int row = half ? r + 8 : r;
      #pragma unroll
      for (int nf = 0; nf < 4; ++nf)
        Psm[wave][row][nf * 16 + l15] = f2bf(p[nf][r]);
    }
    __syncthreads();

    #pragma unroll
    for (int kk = 0; kk < 2; ++kk) {
      FragAB pa;
      {
        int klo = kk * 32 + (half ? 8 : 0);
        pa.q[0] = *(const uint4*)&Psm[wave][l15][klo];
        pa.q[1] = *(const uint4*)&Psm[wave][l15][klo + 16];
      }
      #pragma unroll
      for (int nf = 0; nf < 4; ++nf) {
        FragAB vb;
        int dcol = nf * 16 + l15;
        int klo  = kk * 32 + (half ? 8 : 0);
        vb.q[0] = *(const uint4*)&Vsm[dcol][klo];
        vb.q[1] = *(const uint4*)&Vsm[dcol][klo + 16];
        oacc[nf] = __builtin_amdgcn_wmma_f32_16x16x32_bf16(
            false, pa.v, false, vb.v, (short)0, oacc[nf], false, false);
      }
    }
  }

  // ---- finalize: O /= l, store bf16 into [B,S,E]
  #pragma unroll
  for (int r = 0; r < 8; ++r) {
    int qi = q0 + wave * 16 + (half ? r + 8 : r);
    float inv = 1.0f / lrow[r];
    #pragma unroll
    for (int nf = 0; nf < 4; ++nf) {
      int d = nf * 16 + l15;
      O[base + (size_t)qi * E + d] = f2bf(oacc[nf][r] * inv);
    }
  }
}

// ---------------------------------------------------------------------------
// Host-side orchestration
// ---------------------------------------------------------------------------
extern "C" void kernel_launch(void* const* d_in, const int* in_sizes, int n_in,
                              void* d_out, int out_size, void* d_ws, size_t ws_size,
                              hipStream_t stream) {
  constexpr int Bn = 2, S = 2048, E = 1024, H = 16;
  constexpr int M = Bn * S;                    // 4096
  const float* x  = (const float*)d_in[0];     // [B,S,E]
  const float* aw = (const float*)d_in[1];     // [E,3E]
  const float* ab = (const float*)d_in[2];     // [3E]
  const float* pw = (const float*)d_in[3];     // [E,E]
  const float* pb = (const float*)d_in[4];     // [E]
  float* out = (float*)d_out;                  // [B,S,E] fp32

  // Workspace layout (bf16), ~50 MB total.
  unsigned short* ws   = (unsigned short*)d_ws;
  unsigned short* x_bf = ws;                          // M*E
  unsigned short* wqT  = x_bf + (size_t)M * E;        // E*E  (B^T of Wq)
  unsigned short* wkT  = wqT  + (size_t)E * E;        // E*E  (B^T of Wk^T = Wk)
  unsigned short* wvT  = wkT  + (size_t)E * E;        // E*E
  unsigned short* prT  = wvT  + (size_t)E * E;        // E*E
  unsigned short* tbuf = prT  + (size_t)E * E;        // M*E  x@Wq+qb
  unsigned short* qbuf = tbuf + (size_t)M * E;        // M*E  fused query
  unsigned short* vbuf = qbuf + (size_t)M * E;        // M*E  value proj
  unsigned short* obuf = vbuf + (size_t)M * E;        // M*E  attention output

  cvt_f32_bf16_kernel<<<(M * E + 255) / 256, 256, 0, stream>>>(x, x_bf, M * E);
  cvt_transpose_kernel<<<(E * E + 255) / 256, 256, 0, stream>>>(aw, wqT, E, E, 3 * E, 0);
  cvt_rows_kernel     <<<(E * E + 255) / 256, 256, 0, stream>>>(aw, wkT, E, E, 3 * E, E);
  cvt_transpose_kernel<<<(E * E + 255) / 256, 256, 0, stream>>>(aw, wvT, E, E, 3 * E, 2 * E);
  cvt_transpose_kernel<<<(E * E + 255) / 256, 256, 0, stream>>>(pw, prT, E, E, E, 0);

  dim3 ggrid(E / 128, M / 128);                       // (8, 32)
  gemm_bf16_kernel<false><<<ggrid, 256, 0, stream>>>(x_bf, wqT, ab,         tbuf, nullptr, M, E, E);
  gemm_bf16_kernel<false><<<ggrid, 256, 0, stream>>>(x_bf, wvT, ab + 2 * E, vbuf, nullptr, M, E, E);
  gemm_bf16_kernel<false><<<ggrid, 256, 0, stream>>>(tbuf, wkT, ab + E,     qbuf, nullptr, M, E, E);

  dim3 agrid(S / 128, H, Bn);                         // (16, 16, 2)
  flash_attn_kernel<<<agrid, 256, 0, stream>>>(qbuf, x_bf, vbuf, obuf, S, E, 0.125f);

  gemm_bf16_kernel<true><<<ggrid, 256, 0, stream>>>(obuf, prT, pb, nullptr, out, M, E, E);
}
